// Tokenkenwise_35974646071351
// MI455X (gfx1250) — matrix operand order repaired and checked by reading, not compiled, not run on hardware
//
#include <hip/hip_runtime.h>

// ---------------- problem constants (from reference setup) ----------------
#define B_    8
#define N_    4096
#define IND   512      // in_dims (== D_ so both GEMMs share K=512)
#define D_    512      // TOKEN_DIM * NUM_HEADS
#define TD    256      // TOKEN_DIM
#define M_    (B_ * N_)        // 32768 rows
#define SCALE_ 0.0625f // 256^-0.5

typedef __attribute__((ext_vector_type(16))) __bf16 v16bf;
typedef __attribute__((ext_vector_type(8)))  float  v8f;
typedef __attribute__((ext_vector_type(4)))  unsigned int u32x4;
typedef __attribute__((ext_vector_type(8)))  unsigned int u32x8;

// ---------------- GEMM tiling ----------------
#define GTM 256                        // block rows (8 waves x 32 rows)
#define GTK 32                         // K step
#define GTN 32                         // block cols (2 x 16-wide WMMA tiles)
#define LPITCH 40                      // LDS row pitch in bf16 (80 B, 16B aligned; TDM pad)
#define ASB (GTM * LPITCH * 2)         // 20480 B per A buffer
#define BSB (GTN * LPITCH * 2)         // 2560 B per B buffer
#define SMEM_BYTES (2 * ASB + 2 * BSB) // double-buffered

// ---------------- small helpers ----------------
__device__ __forceinline__ float blk_sum(float v, float* red, int t, int nt) {
  red[t] = v; __syncthreads();
  for (int off = nt >> 1; off > 0; off >>= 1) {
    if (t < off) red[t] += red[t + off];
    __syncthreads();
  }
  float r = red[0]; __syncthreads();
  return r;
}
__device__ __forceinline__ float blk_max(float v, float* red, int t, int nt) {
  red[t] = v; __syncthreads();
  for (int off = nt >> 1; off > 0; off >>= 1) {
    if (t < off) red[t] = fmaxf(red[t], red[t + off]);
    __syncthreads();
  }
  float r = red[0]; __syncthreads();
  return r;
}

// convert + transpose: in [K][Ncols] f32 row-major -> out [Ncols][K] bf16 row-major
__global__ void to_bf16_tr_k(const float* __restrict__ in, __bf16* __restrict__ out,
                             int K, int Ncols) {
  int idx = blockIdx.x * blockDim.x + threadIdx.x;
  if (idx < K * Ncols) {
    int k = idx / Ncols, n = idx - k * Ncols;
    out[(size_t)n * K + k] = (__bf16)in[idx];
  }
}

// bulk f32 -> bf16 (vectorized: float4 in, 4 packed bf16 out)
__global__ void f32_to_bf16_v4_k(const float* __restrict__ in, __bf16* __restrict__ out, int n4) {
  int i = blockIdx.x * blockDim.x + threadIdx.x;
  if (i < n4) {
    float4 v = ((const float4*)in)[i];
    union { __bf16 h[4]; uint2 u; } p;
    p.h[0] = (__bf16)v.x; p.h[1] = (__bf16)v.y; p.h[2] = (__bf16)v.z; p.h[3] = (__bf16)v.w;
    ((uint2*)out)[i] = p.u;
  }
}

// ---------------- Tensor Data Mover: 2D bf16 tile -> LDS (padded rows) ----------------
// Tile (tile_d1 rows x 32 K-elements) from a [d1_total][d0_total] bf16 tensor,
// row stride d0_stride elements. LDS rows padded to 80B: pad 4 DWORDs every 16 DWORDs.
__device__ __forceinline__ void tdm_load_tile_bf16(unsigned ldsOff, const __bf16* gtile,
                                                   unsigned d0_total, unsigned d1_total,
                                                   unsigned d0_stride, unsigned tile_d1) {
  unsigned long long ga = (unsigned long long)(uintptr_t)gtile;
  u32x4 g0;
  g0[0] = 1u;                                       // count=1, user descriptor
  g0[1] = ldsOff;                                   // lds_addr
  g0[2] = (unsigned)(ga & 0xffffffffu);             // global_addr[31:0]
  g0[3] = (unsigned)((ga >> 32) & 0x1ffffffu)       // global_addr[56:32]
          | 0x80000000u;                            // type=2 ("image")
  u32x8 g1;
  g1[0] = 0x00010000u                               // data_size=1 (2B)
          | (1u << 20)                              // pad_enable
          | (3u << 22)                              // pad_interval: 16 DWORDs (64B)
          | (3u << 25);                             // pad_amount: 4 DWORDs (16B)
  g1[1] = (d0_total & 0xffffu) << 16;               // tensor_dim0[15:0]
  g1[2] = ((d0_total >> 16) & 0xffffu)
          | ((d1_total & 0xffffu) << 16);           // tensor_dim1[15:0]
  g1[3] = ((d1_total >> 16) & 0xffffu)
          | ((unsigned)GTK << 16);                  // tile_dim0 = 32
  g1[4] = tile_d1;                                  // tile_dim1 (tile_dim2=0)
  g1[5] = d0_stride;                                // tensor_dim0_stride[31:0]
  g1[6] = 0u;
  g1[7] = 0u;
  asm volatile("tensor_load_to_lds %0, %1" :: "s"(g0), "s"(g1) : "memory");
}

union frag16 { v16bf v; uint4 q[2]; };

// ---------------- generic TDM-fed GEMM: out[M][Ntot] = Abf[M][512] x WT[Ntot][512]^T + bias ----
__global__ void __launch_bounds__(256)
gemm_tdm_wmma(const __bf16* __restrict__ Abf, const __bf16* __restrict__ WT,
              const float* __restrict__ bias, float* __restrict__ out, int Ntot) {
  extern __shared__ char smem[];
  const unsigned gss = __builtin_amdgcn_groupstaticsize();
  const unsigned asOff[2] = {gss, gss + ASB};
  const unsigned bsOff[2] = {gss + 2 * ASB, gss + 2 * ASB + BSB};

  const int t    = threadIdx.x;
  const int lane = t & 31;
  const int w    = t >> 5;           // wave id 0..7
  const int hi   = lane >> 4;
  const int lo   = lane & 15;
  const int row0 = blockIdx.y * GTM;
  const int col0 = blockIdx.x * GTN;

  const __bf16* Ab = Abf + (size_t)row0 * IND;   // A tile row start
  const __bf16* Bb = WT + (size_t)col0 * IND;    // B tile row start (K-contiguous)

  v8f acc00 = {}, acc01 = {}, acc10 = {}, acc11 = {};

  // prologue: DMA first tiles into buffer 0
  if (w == 0) {
    tdm_load_tile_bf16(asOff[0], Ab, IND, M_, IND, GTM);
    tdm_load_tile_bf16(bsOff[0], Bb, IND, (unsigned)Ntot, IND, GTN);
  }

  #pragma unroll 2
  for (int kt = 0; kt < IND; kt += GTK) {
    const int cur = (kt >> 5) & 1;
    const int nxt = cur ^ 1;
    if (w == 0) {
      if (kt + GTK < IND) {
        // pipeline next tiles into the other buffer, then wait for current pair
        tdm_load_tile_bf16(asOff[nxt], Ab + kt + GTK, IND, M_, IND, GTM);
        tdm_load_tile_bf16(bsOff[nxt], Bb + kt + GTK, IND, (unsigned)Ntot, IND, GTN);
        __builtin_amdgcn_s_wait_tensorcnt(2);
      } else {
        __builtin_amdgcn_s_wait_tensorcnt(0);
      }
    }
    __syncthreads();

    const __bf16* Ac = (const __bf16*)(smem + cur * ASB);
    const __bf16* Bc = (const __bf16*)(smem + 2 * ASB + cur * BSB);

    frag16 a0, a1, b0, b1;
    const uint4* ar0 = (const uint4*)(Ac + (size_t)(w * 32 + lo) * LPITCH);
    const uint4* ar1 = (const uint4*)(Ac + (size_t)(w * 32 + 16 + lo) * LPITCH);
    a0.q[0] = ar0[hi];      a0.q[1] = ar0[2 + hi];
    a1.q[0] = ar1[hi];      a1.q[1] = ar1[2 + hi];
    const uint4* br0 = (const uint4*)(Bc + (size_t)lo * LPITCH);
    const uint4* br1 = (const uint4*)(Bc + (size_t)(16 + lo) * LPITCH);
    b0.q[0] = br0[hi];      b0.q[1] = br0[2 + hi];
    b1.q[0] = br1[hi];      b1.q[1] = br1[2 + hi];

    acc00 = __builtin_amdgcn_wmma_f32_16x16x32_bf16(false, a0.v, false, b0.v, (short)0, acc00, false, false);
    acc01 = __builtin_amdgcn_wmma_f32_16x16x32_bf16(false, a0.v, false, b1.v, (short)0, acc01, false, false);
    acc10 = __builtin_amdgcn_wmma_f32_16x16x32_bf16(false, a1.v, false, b0.v, (short)0, acc10, false, false);
    acc11 = __builtin_amdgcn_wmma_f32_16x16x32_bf16(false, a1.v, false, b1.v, (short)0, acc11, false, false);
    __syncthreads();   // all reads of buf[cur] done before it is refilled next iter
  }

  #pragma unroll
  for (int r = 0; r < 8; ++r) {
    int m0 = row0 + w * 32 + r + 8 * hi;
    int m1 = m0 + 16;
    int c0 = col0 + lo;
    out[(size_t)m0 * Ntot + c0]      = acc00[r] + bias[c0];
    out[(size_t)m0 * Ntot + c0 + 16] = acc01[r] + bias[c0 + 16];
    out[(size_t)m1 * Ntot + c0]      = acc10[r] + bias[c0];
    out[(size_t)m1 * Ntot + c0 + 16] = acc11[r] + bias[c0 + 16];
  }
}

// ---------------- row l2norm of q,k + alpha_raw ----------------
__global__ void rownorm_alpha_k(float* __restrict__ q, float* __restrict__ k,
                                const float* __restrict__ w_alpha, float* __restrict__ alpha) {
  const int row = blockIdx.x;
  const int t   = threadIdx.x;      // 128
  __shared__ float red[128];
  float* qr = q + (size_t)row * D_;
  float* kr = k + (size_t)row * D_;

  float s = 0.f;
  for (int i = t; i < D_; i += 128) { float v = qr[i]; s += v * v; }
  s = blk_sum(s, red, t, 128);
  float inv = 1.0f / fmaxf(sqrtf(s), 1e-12f);

  float a = 0.f;
  for (int i = t; i < D_; i += 128) { float v = qr[i] * inv; qr[i] = v; a += v * w_alpha[i]; }
  a = blk_sum(a, red, t, 128);
  if (t == 0) alpha[row] = a * SCALE_;

  s = 0.f;
  for (int i = t; i < D_; i += 128) { float v = kr[i]; s += v * v; }
  s = blk_sum(s, red, t, 128);
  inv = 1.0f / fmaxf(sqrtf(s), 1e-12f);   // v = normalize(normalize(k)) == normalize(k)
  for (int i = t; i < D_; i += 128) kr[i] *= inv;
}

// ---------------- normalize a [B_,N_] vector over n ----------------
__global__ void vecnorm_n_k(float* __restrict__ a) {
  const int b = blockIdx.x, t = threadIdx.x;  // 256
  __shared__ float red[256];
  float* ab = a + (size_t)b * N_;
  float s = 0.f;
  for (int i = t; i < N_; i += 256) { float v = ab[i]; s += v * v; }
  s = blk_sum(s, red, t, 256);
  float inv = 1.0f / fmaxf(sqrtf(s), 1e-12f);
  for (int i = t; i < N_; i += 256) ab[i] *= inv;
}

// ---------------- g_q = softmax_d( sum_n alpha*q );  gw = g_q * w_beta ----------------
__global__ void __launch_bounds__(512)
gq_softmax_k(const float* __restrict__ alpha, const float* __restrict__ q,
             const float* __restrict__ w_beta, float* __restrict__ g_q, float* __restrict__ gw) {
  const int b = blockIdx.x, d = threadIdx.x;  // 512 == D_
  __shared__ float red[512];
  const float* qb = q + (size_t)b * N_ * D_;
  const float* ab = alpha + (size_t)b * N_;
  float s = 0.f;
  for (int n = 0; n < N_; ++n) s += ab[n] * qb[(size_t)n * D_ + d];
  float m  = blk_max(s, red, d, 512);
  float e  = expf(s - m);
  float sm = blk_sum(e, red, d, 512);
  float g  = e / sm;
  g_q[(size_t)b * D_ + d] = g;
  gw[(size_t)b * D_ + d]  = g * w_beta[d];
}

// ---------------- beta_raw[b,n] = SCALE * k_hat . gw ----------------
__global__ void beta_raw_k(const float* __restrict__ k, const float* __restrict__ gw,
                           float* __restrict__ beta) {
  const int row = blockIdx.x;
  const int b   = row / N_;
  const int t   = threadIdx.x;   // 128
  __shared__ float red[128];
  const float* kr = k + (size_t)row * D_;
  const float* g  = gw + (size_t)b * D_;
  float s = 0.f;
  for (int i = t; i < D_; i += 128) s += kr[i] * g[i];
  s = blk_sum(s, red, t, 128);
  if (t == 0) beta[row] = s * SCALE_;
}

// ---------------- g_k = softmax_d( g_q * sum_n beta*k ) ----------------
__global__ void __launch_bounds__(512)
gk_softmax_k(const float* __restrict__ beta, const float* __restrict__ k,
             const float* __restrict__ g_q, float* __restrict__ g_k) {
  const int b = blockIdx.x, d = threadIdx.x;
  __shared__ float red[512];
  const float* kb = k + (size_t)b * N_ * D_;
  const float* bb = beta + (size_t)b * N_;
  float s = 0.f;
  for (int n = 0; n < N_; ++n) s += bb[n] * kb[(size_t)n * D_ + d];
  s *= g_q[(size_t)b * D_ + d];
  float m  = blk_max(s, red, d, 512);
  float e  = expf(s - m);
  float sm = blk_sum(e, red, d, 512);
  g_k[(size_t)b * D_ + d] = e / sm;
}

// ---------------- kvi = g_k*k_hat + q_hat  -> bf16 (A matrix of final GEMM) ----------------
__global__ void kvi_bf16_k(const float* __restrict__ q, const float* __restrict__ k,
                           const float* __restrict__ g_k, __bf16* __restrict__ out) {
  int i = blockIdx.x * blockDim.x + threadIdx.x;   // float4 index, D_/4=128 per row
  int row = i >> 7;
  int c4  = i & 127;
  int b   = row >> 12;                             // row / N_
  float4 kv = ((const float4*)k)[i];
  float4 qv = ((const float4*)q)[i];
  float4 gv = ((const float4*)(g_k + (size_t)b * D_))[c4];
  union { __bf16 h[4]; uint2 u; } p;
  p.h[0] = (__bf16)(gv.x * kv.x + qv.x);
  p.h[1] = (__bf16)(gv.y * kv.y + qv.y);
  p.h[2] = (__bf16)(gv.z * kv.z + qv.z);
  p.h[3] = (__bf16)(gv.w * kv.w + qv.w);
  ((uint2*)out)[i] = p.u;
}

// ---------------- host side ----------------
extern "C" void kernel_launch(void* const* d_in, const int* in_sizes, int n_in,
                              void* d_out, int out_size, void* d_ws, size_t ws_size,
                              hipStream_t stream) {
  const float* x       = (const float*)d_in[0];
  const float* Wq      = (const float*)d_in[1];
  const float* bq      = (const float*)d_in[2];
  const float* Wk      = (const float*)d_in[3];
  const float* bk      = (const float*)d_in[4];
  const float* Wr      = (const float*)d_in[5];
  const float* br      = (const float*)d_in[6];
  const float* w_alpha = (const float*)d_in[7];
  const float* w_beta  = (const float*)d_in[8];
  float* out = (float*)d_out;

  // workspace layout
  char* ws = (char*)d_ws;
  float*  qbuf   = (float*)ws;                                   // 64 MB
  float*  kbuf   = (float*)(ws + (size_t)67108864);              // 64 MB
  __bf16* x_bf   = (__bf16*)(ws + (size_t)134217728);            // 32 MB  (later reused for kvi)
  __bf16* WqT_bf = (__bf16*)(ws + (size_t)167772160);            // 512 KB [N][K]
  __bf16* WkT_bf = WqT_bf + 512 * 512;                           // 512 KB
  __bf16* WrT_bf = WkT_bf + 512 * 512;                           // 256 KB [256][512]
  float*  alpha  = (float*)(WrT_bf + 256 * 512);                 // 128 KB
  float*  beta   = alpha + B_ * N_;                              // 128 KB
  float*  g_q    = beta + B_ * N_;                               // 16 KB
  float*  gw     = g_q + B_ * D_;                                // 16 KB
  float*  g_k    = gw + B_ * D_;                                 // 16 KB
  __bf16* kvi_bf = x_bf;                                         // alias: x dead after q/k GEMMs

  to_bf16_tr_k<<<(512 * 512 + 255) / 256, 256, 0, stream>>>(Wq, WqT_bf, 512, 512);
  to_bf16_tr_k<<<(512 * 512 + 255) / 256, 256, 0, stream>>>(Wk, WkT_bf, 512, 512);
  to_bf16_tr_k<<<(512 * 256 + 255) / 256, 256, 0, stream>>>(Wr, WrT_bf, 512, 256);
  f32_to_bf16_v4_k<<<(M_ * IND / 4 + 255) / 256, 256, 0, stream>>>(x, x_bf, M_ * IND / 4);

  dim3 g1(D_ / GTN, M_ / GTM);                       // (16, 128)
  gemm_tdm_wmma<<<g1, 256, SMEM_BYTES, stream>>>(x_bf, WqT_bf, bq, qbuf, D_);
  gemm_tdm_wmma<<<g1, 256, SMEM_BYTES, stream>>>(x_bf, WkT_bf, bk, kbuf, D_);

  rownorm_alpha_k<<<M_, 128, 0, stream>>>(qbuf, kbuf, w_alpha, alpha);
  vecnorm_n_k<<<B_, 256, 0, stream>>>(alpha);
  gq_softmax_k<<<B_, 512, 0, stream>>>(alpha, qbuf, w_beta, g_q, gw);
  beta_raw_k<<<M_, 128, 0, stream>>>(kbuf, gw, beta);
  vecnorm_n_k<<<B_, 256, 0, stream>>>(beta);
  gk_softmax_k<<<B_, 512, 0, stream>>>(beta, kbuf, g_q, g_k);

  kvi_bf16_k<<<(M_ * D_ / 4 + 255) / 256, 256, 0, stream>>>(qbuf, kbuf, g_k, kvi_bf);

  dim3 g2(TD / GTN, M_ / GTM);                       // (8, 128)
  gemm_tdm_wmma<<<g2, 256, SMEM_BYTES, stream>>>(kvi_bf, WrT_bf, br, out, TD);
}
